// GraphMETNetwork_55319178772885
// MI455X (gfx1250) — matrix operand order repaired and checked by reading, compiled, not run on hardware
//
#include <hip/hip_runtime.h>
#include <hip/hip_bf16.h>
#include <stdint.h>

typedef __attribute__((ext_vector_type(16))) _Float16 v16h;
typedef __attribute__((ext_vector_type(8)))  _Float16 v8h;
typedef __attribute__((ext_vector_type(8)))  float    v8f;

#define ENC_NEGINF 0x007FFFFFu

// ---------------------------------------------------------------------------
// WMMA tile loaders (CDNA5 16x16x32 f16 layouts, wave32)
// A 16x32: lane l (h=l>>4, m=l&15) holds row m, K = kbase + 8h + (e<8 ? e : e+8)
// B 32x16 (pre-packed): lane l holds col n=l&15, K = kbase + 8*(l>>4) + (e<8 ? e : e+8),
//   packed contiguously: 16 f16 per lane per tile.
// C/D f32: reg r, lane l -> M = r + 8*(l>>4), N = l&15
// ---------------------------------------------------------------------------
static __device__ inline v16h load_a_tile(const _Float16* base, int rowStride,
                                          int kbase, int lane) {
  const int h = lane >> 4;
  const int m = lane & 15;
  const _Float16* rp = base + m * rowStride + kbase + 8 * h;
  v16h a;
#pragma unroll
  for (int e = 0; e < 16; ++e) {
    int k = (e < 8) ? e : (e + 8);   // two contiguous 8-f16 spans -> 2x ds_load_b128
    a[e] = rp[k];
  }
  return a;
}

static __device__ inline v16h load_b_tile(const _Float16* packed, int tile, int lane) {
  return *(const v16h*)(packed + (size_t)(tile * 32 + lane) * 16);
}

static __device__ inline uint32_t encode_f32(float v) {
  uint32_t b = __float_as_uint(v);
  return (b & 0x80000000u) ? ~b : (b | 0x80000000u);
}
static __device__ inline float decode_f32(uint32_t e) {
  uint32_t b = (e & 0x80000000u) ? (e & 0x7FFFFFFFu) : ~e;
  return __uint_as_float(b);
}

// ---------------------------------------------------------------------------
// Weight packing: W[K x Nw] f32 -> B-tiles, K zero-padded to Kp (mult of 32)
// ---------------------------------------------------------------------------
__global__ void pack_weight_kernel(const float* __restrict__ W, int K, int Nw,
                                   int Kp, _Float16* __restrict__ dst) {
  int ntilesN = Nw >> 4;
  int total = (Kp >> 5) * ntilesN * 512;
  for (int idx = blockIdx.x * blockDim.x + threadIdx.x; idx < total;
       idx += gridDim.x * blockDim.x) {
    int tile = idx >> 9;
    int within = idx & 511;
    int l = within >> 4;
    int e = within & 15;
    int kt = tile / ntilesN, nt = tile % ntilesN;
    int srcK = kt * 32 + ((l >> 4) << 3) + ((e < 8) ? e : (e + 8));
    int col = nt * 16 + (l & 15);
    float v = (srcK < K) ? W[srcK * Nw + col] : 0.0f;
    dst[idx] = (_Float16)v;
  }
}

__global__ void fill_u32_kernel(uint32_t* __restrict__ p, uint32_t v, int n) {
  int i = blockIdx.x * blockDim.x + threadIdx.x;
  if (i < n) p[i] = v;
}

// ---------------------------------------------------------------------------
// Node-embedding MLP: x[N x F] -> 3 layers (F->64 relu, 64->64 relu, 64->64)
// ---------------------------------------------------------------------------
__global__ __launch_bounds__(256) void emb_mlp_kernel(
    const float* __restrict__ x, int F, int N,
    const _Float16* __restrict__ W1p, const float* __restrict__ b1,
    const _Float16* __restrict__ W2p, const float* __restrict__ b2,
    const _Float16* __restrict__ W3p, const float* __restrict__ b3,
    float* __restrict__ emb32, _Float16* __restrict__ emb16) {
  __shared__ _Float16 smem[8][16 * 64];
  const int wave = threadIdx.x >> 5;
  const int lane = threadIdx.x & 31;
  _Float16* hs = smem[wave];
  const int m0 = (blockIdx.x * 8 + wave) * 16;
  const int hLane = lane >> 4, nLane = lane & 15;

  // stage x as [16 x 32] f16 (zero-pad cols >= F), row stride 32
  for (int idx = lane; idx < 16 * 32; idx += 32) {
    int r = idx >> 5, c = idx & 31;
    int row = m0 + r; if (row >= N) row = N - 1;
    float v = (c < F) ? x[row * F + c] : 0.0f;
    hs[r * 32 + c] = (_Float16)v;
  }
  v16h a0 = load_a_tile(hs, 32, 0, lane);

  v8f acc[4];
  // layer 1: K=32 (padded), N=64
#pragma unroll
  for (int nt = 0; nt < 4; ++nt) {
    v8f c = {};
    c = __builtin_amdgcn_wmma_f32_16x16x32_f16(false, a0, false,
          load_b_tile(W1p, nt, lane), (short)0, c, false, false);
    acc[nt] = c;
  }
#pragma unroll
  for (int nt = 0; nt < 4; ++nt) {
    int col = nt * 16 + nLane;
    float bb = b1[col];
#pragma unroll
    for (int r = 0; r < 8; ++r) {
      float v = fmaxf(acc[nt][r] + bb, 0.0f);
      hs[(r + 8 * hLane) * 64 + col] = (_Float16)v;
    }
  }
  // layer 2: K=64, N=64
  {
    v16h A0 = load_a_tile(hs, 64, 0, lane);
    v16h A1 = load_a_tile(hs, 64, 32, lane);
#pragma unroll
    for (int nt = 0; nt < 4; ++nt) {
      v8f c = {};
      c = __builtin_amdgcn_wmma_f32_16x16x32_f16(false, A0, false,
            load_b_tile(W2p, nt, lane), (short)0, c, false, false);
      c = __builtin_amdgcn_wmma_f32_16x16x32_f16(false, A1, false,
            load_b_tile(W2p, 4 + nt, lane), (short)0, c, false, false);
      acc[nt] = c;
    }
#pragma unroll
    for (int nt = 0; nt < 4; ++nt) {
      int col = nt * 16 + nLane;
      float bb = b2[col];
#pragma unroll
      for (int r = 0; r < 8; ++r) {
        float v = fmaxf(acc[nt][r] + bb, 0.0f);
        hs[(r + 8 * hLane) * 64 + col] = (_Float16)v;
      }
    }
  }
  // layer 3: K=64, N=64, no relu; write f32 + f16 outputs
  {
    v16h A0 = load_a_tile(hs, 64, 0, lane);
    v16h A1 = load_a_tile(hs, 64, 32, lane);
#pragma unroll
    for (int nt = 0; nt < 4; ++nt) {
      v8f c = {};
      c = __builtin_amdgcn_wmma_f32_16x16x32_f16(false, A0, false,
            load_b_tile(W3p, nt, lane), (short)0, c, false, false);
      c = __builtin_amdgcn_wmma_f32_16x16x32_f16(false, A1, false,
            load_b_tile(W3p, 4 + nt, lane), (short)0, c, false, false);
      int col = nt * 16 + nLane;
      float bb = b3[col];
#pragma unroll
      for (int r = 0; r < 8; ++r) {
        int row = m0 + r + 8 * hLane;
        if (row < N) {
          float v = c[r] + bb;
          emb32[(size_t)row * 64 + col] = v;
          emb16[(size_t)row * 64 + col] = (_Float16)v;
        }
      }
    }
  }
}

// ---------------------------------------------------------------------------
// EdgeConv: m = MLP([x_i, x_j - x_i]) (128->96 relu, 96->64); max-agg by dst
// ---------------------------------------------------------------------------
__global__ __launch_bounds__(256) void edge_conv_kernel(
    const _Float16* __restrict__ emb16,
    const int* __restrict__ esrc, const int* __restrict__ edst,
    const _Float16* __restrict__ W1p, const float* __restrict__ b1,
    const _Float16* __restrict__ W2p, const float* __restrict__ b2,
    uint32_t* __restrict__ agg, int E, int N) {
  __shared__ _Float16 smem[8][16 * 128];
  __shared__ int sdst[8][16];
  const int wave = threadIdx.x >> 5;
  const int lane = threadIdx.x & 31;
  _Float16* am = smem[wave];
  int* dw = sdst[wave];
  const int e0 = (blockIdx.x * 8 + wave) * 16;
  const int hLane = lane >> 4, nLane = lane & 15;

  // gather: 2 lanes per edge row; cols [0..63]=x_i, [64..127]=x_j - x_i
  // vectorized: v8h global loads + packed f16 subtract + ds_store_b128
  {
    int r = lane >> 1, hh = lane & 1;
    int e = e0 + r; if (e >= E) e = E - 1;
    int s = esrc[e], d = edst[e];
    if (hh == 0) dw[r] = d;
    const _Float16* pi = emb16 + (size_t)d * 64 + hh * 32;
    const _Float16* pj = emb16 + (size_t)s * 64 + hh * 32;
    _Float16* oi = am + r * 128 + hh * 32;
#pragma unroll
    for (int c = 0; c < 4; ++c) {
      v8h xi = *(const v8h*)(pi + c * 8);
      v8h xj = *(const v8h*)(pj + c * 8);
      *(v8h*)(oi + c * 8)      = xi;        // ds_store_b128
      *(v8h*)(oi + 64 + c * 8) = xj - xi;   // v_pk_sub_f16 + ds_store_b128
    }
  }

  v16h a[4];
#pragma unroll
  for (int kt = 0; kt < 4; ++kt) a[kt] = load_a_tile(am, 128, kt * 32, lane);

  // layer 1: [16,128] x [128,96], relu -> am as [16 x 96]
#pragma unroll
  for (int nt = 0; nt < 6; ++nt) {
    v8f c = {};
#pragma unroll
    for (int kt = 0; kt < 4; ++kt)
      c = __builtin_amdgcn_wmma_f32_16x16x32_f16(false, a[kt], false,
            load_b_tile(W1p, kt * 6 + nt, lane), (short)0, c, false, false);
    int col = nt * 16 + nLane;
    float bb = b1[col];
#pragma unroll
    for (int r = 0; r < 8; ++r) {
      float v = fmaxf(c[r] + bb, 0.0f);
      am[(r + 8 * hLane) * 96 + col] = (_Float16)v;
    }
  }

  v16h a2[3];
#pragma unroll
  for (int kt = 0; kt < 3; ++kt) a2[kt] = load_a_tile(am, 96, kt * 32, lane);

  // layer 2: [16,96] x [96,64] -> atomic segment-max
#pragma unroll
  for (int nt = 0; nt < 4; ++nt) {
    v8f c = {};
#pragma unroll
    for (int kt = 0; kt < 3; ++kt)
      c = __builtin_amdgcn_wmma_f32_16x16x32_f16(false, a2[kt], false,
            load_b_tile(W2p, kt * 4 + nt, lane), (short)0, c, false, false);
    int col = nt * 16 + nLane;
    float bb = b2[col];
#pragma unroll
    for (int r = 0; r < 8; ++r) {
      int M = r + 8 * hLane;
      int e = e0 + M;
      if (e < E) {
        uint32_t enc = encode_f32(c[r] + bb);
        atomicMax(&agg[(size_t)dw[M] * 64 + col], enc);
      }
    }
  }
}

// ---------------------------------------------------------------------------
// Finalize: emb += max-agg (-inf -> 0), concat [cont|cat] -> out MLP
// (128->64 relu, 64->32 relu, 32->1)
// ---------------------------------------------------------------------------
__global__ __launch_bounds__(256) void out_mlp_kernel(
    const float* __restrict__ embC32, const uint32_t* __restrict__ aggC,
    const float* __restrict__ embK32, const uint32_t* __restrict__ aggK,
    const _Float16* __restrict__ W1p, const float* __restrict__ b1,
    const _Float16* __restrict__ W2p, const float* __restrict__ b2,
    const float* __restrict__ W3, const float* __restrict__ b3,
    float* __restrict__ out, int N) {
  __shared__ _Float16 amem[8][16 * 128];
  __shared__ float h2mem[8][16 * 32];
  const int wave = threadIdx.x >> 5;
  const int lane = threadIdx.x & 31;
  _Float16* am = amem[wave];
  float* h2 = h2mem[wave];
  const int m0 = (blockIdx.x * 8 + wave) * 16;
  const int hLane = lane >> 4, nLane = lane & 15;

  // build concat features: hh=0 -> cont cols 0..63, hh=1 -> cat cols 64..127
  // vectorized into v8h chunks -> ds_store_b128; global loads merge to b128
  {
    int r = lane >> 1, hh = lane & 1;
    int row = m0 + r; if (row >= N) row = N - 1;
    const float* e32 = hh ? embK32 : embC32;
    const uint32_t* ag = hh ? aggK : aggC;
    _Float16* od = am + r * 128 + hh * 64;
#pragma unroll
    for (int c8 = 0; c8 < 8; ++c8) {
      v8h o;
#pragma unroll
      for (int k = 0; k < 8; ++k) {
        int c = c8 * 8 + k;
        uint32_t e = ag[(size_t)row * 64 + c];
        float mv = (e == ENC_NEGINF) ? 0.0f : decode_f32(e);
        o[k] = (_Float16)(e32[(size_t)row * 64 + c] + mv);
      }
      *(v8h*)(od + c8 * 8) = o;
    }
  }
  v16h a[4];
#pragma unroll
  for (int kt = 0; kt < 4; ++kt) a[kt] = load_a_tile(am, 128, kt * 32, lane);

  // layer 1: 128 -> 64, relu -> am as [16 x 64]
#pragma unroll
  for (int nt = 0; nt < 4; ++nt) {
    v8f c = {};
#pragma unroll
    for (int kt = 0; kt < 4; ++kt)
      c = __builtin_amdgcn_wmma_f32_16x16x32_f16(false, a[kt], false,
            load_b_tile(W1p, kt * 4 + nt, lane), (short)0, c, false, false);
    int col = nt * 16 + nLane;
    float bb = b1[col];
#pragma unroll
    for (int r = 0; r < 8; ++r) {
      float v = fmaxf(c[r] + bb, 0.0f);
      am[(r + 8 * hLane) * 64 + col] = (_Float16)v;
    }
  }
  // layer 2: 64 -> 32, relu -> h2 (f32)
  {
    v16h A0 = load_a_tile(am, 64, 0, lane);
    v16h A1 = load_a_tile(am, 64, 32, lane);
#pragma unroll
    for (int nt = 0; nt < 2; ++nt) {
      v8f c = {};
      c = __builtin_amdgcn_wmma_f32_16x16x32_f16(false, A0, false,
            load_b_tile(W2p, nt, lane), (short)0, c, false, false);
      c = __builtin_amdgcn_wmma_f32_16x16x32_f16(false, A1, false,
            load_b_tile(W2p, 2 + nt, lane), (short)0, c, false, false);
      int col = nt * 16 + nLane;
      float bb = b2[col];
#pragma unroll
      for (int r = 0; r < 8; ++r) {
        float v = fmaxf(c[r] + bb, 0.0f);
        h2[(r + 8 * hLane) * 32 + col] = v;
      }
    }
  }
  // layer 3: 32 -> 1 (scalar f32)
  if (lane < 16) {
    int row = m0 + lane;
    if (row < N) {
      float s = b3[0];
#pragma unroll
      for (int c = 0; c < 32; ++c) s += h2[lane * 32 + c] * W3[c];
      out[row] = s;
    }
  }
}

// ---------------------------------------------------------------------------
// host launcher
// ---------------------------------------------------------------------------
extern "C" void kernel_launch(void* const* d_in, const int* in_sizes, int n_in,
                              void* d_out, int out_size, void* d_ws, size_t ws_size,
                              hipStream_t stream) {
  const int N = 50000, E = 800000;

  // locate inputs by unique flat sizes (robust to pytree key ordering)
  int i_embc = -1, i_embk = -1, i_cv1 = -1, i_cv2 = -1, i_out = -1;
  int i_xc = -1, i_xk = -1, i_ei = -1;
  for (int i = 0; i < n_in; ++i) {
    int s = in_sizes[i];
    if (s == 512 && i_embc < 0) i_embc = i;            // emb_cont W1 (8x64)
    else if (s == 192 && i_embk < 0) i_embk = i;       // emb_cat  W1 (3x64)
    else if (s == 12288) { if (i_cv1 < 0) i_cv1 = i; else if (i_cv2 < 0) i_cv2 = i; }
    else if (s == 8192 && i_out < 0) i_out = i;        // out W1 (128x64)
    else if (s == N * 8) i_xc = i;
    else if (s == N * 3) i_xk = i;
    else if (s == 2 * E) i_ei = i;
  }
  // conv_cont vs conv_cat: flattening preserves relative order of emb_cont/emb_cat
  int i_cvc, i_cvk;
  if (i_embc < i_embk) { i_cvc = i_cv1; i_cvk = i_cv2; }   // insertion order
  else                 { i_cvk = i_cv1; i_cvc = i_cv2; }   // sorted-key order

  auto P = [&](int i) { return (const float*)d_in[i]; };
  const float *Wc1 = P(i_embc), *bc1 = P(i_embc + 1), *Wc2 = P(i_embc + 2),
              *bc2 = P(i_embc + 3), *Wc3 = P(i_embc + 4), *bc3 = P(i_embc + 5);
  const float *Wk1 = P(i_embk), *bk1 = P(i_embk + 1), *Wk2 = P(i_embk + 2),
              *bk2 = P(i_embk + 3), *Wk3 = P(i_embk + 4), *bk3 = P(i_embk + 5);
  const float *WcvC1 = P(i_cvc), *bcvC1 = P(i_cvc + 1), *WcvC2 = P(i_cvc + 2), *bcvC2 = P(i_cvc + 3);
  const float *WcvK1 = P(i_cvk), *bcvK1 = P(i_cvk + 1), *WcvK2 = P(i_cvk + 2), *bcvK2 = P(i_cvk + 3);
  const float *Wo1 = P(i_out), *bo1 = P(i_out + 1), *Wo2 = P(i_out + 2),
              *bo2 = P(i_out + 3), *Wo3 = P(i_out + 4), *bo3 = P(i_out + 5);
  const float* x_cont = P(i_xc);
  const float* x_cat  = P(i_xk);
  const int* ei = (const int*)d_in[i_ei];

  // workspace carve-out
  char* ws = (char*)d_ws;
  size_t off = 0;
  auto carve = [&](size_t bytes) {
    void* p = ws + off;
    off = (off + bytes + 255) & ~(size_t)255;
    return p;
  };
  _Float16* pEc1 = (_Float16*)carve(1 * 4 * 512 * 2);   // K32pad x 64
  _Float16* pEc2 = (_Float16*)carve(2 * 4 * 512 * 2);   // 64 x 64
  _Float16* pEc3 = (_Float16*)carve(2 * 4 * 512 * 2);
  _Float16* pEk1 = (_Float16*)carve(1 * 4 * 512 * 2);
  _Float16* pEk2 = (_Float16*)carve(2 * 4 * 512 * 2);
  _Float16* pEk3 = (_Float16*)carve(2 * 4 * 512 * 2);
  _Float16* pCc1 = (_Float16*)carve(4 * 6 * 512 * 2);   // 128 x 96
  _Float16* pCc2 = (_Float16*)carve(3 * 4 * 512 * 2);   // 96 x 64
  _Float16* pCk1 = (_Float16*)carve(4 * 6 * 512 * 2);
  _Float16* pCk2 = (_Float16*)carve(3 * 4 * 512 * 2);
  _Float16* pO1  = (_Float16*)carve(4 * 4 * 512 * 2);   // 128 x 64
  _Float16* pO2  = (_Float16*)carve(2 * 2 * 512 * 2);   // 64 x 32
  float*     embC32 = (float*)carve((size_t)N * 64 * 4);
  float*     embK32 = (float*)carve((size_t)N * 64 * 4);
  _Float16*  embC16 = (_Float16*)carve((size_t)N * 64 * 2);
  _Float16*  embK16 = (_Float16*)carve((size_t)N * 64 * 2);
  uint32_t*  aggC   = (uint32_t*)carve((size_t)N * 64 * 4);
  uint32_t*  aggK   = (uint32_t*)carve((size_t)N * 64 * 4);
  (void)ws_size;

  auto pack = [&](const float* W, int K, int Nw, int Kp, _Float16* dst) {
    int total = (Kp / 32) * (Nw / 16) * 512;
    pack_weight_kernel<<<(total + 255) / 256, 256, 0, stream>>>(W, K, Nw, Kp, dst);
  };
  pack(Wc1, 8, 64, 32, pEc1);   pack(Wc2, 64, 64, 64, pEc2);  pack(Wc3, 64, 64, 64, pEc3);
  pack(Wk1, 3, 64, 32, pEk1);   pack(Wk2, 64, 64, 64, pEk2);  pack(Wk3, 64, 64, 64, pEk3);
  pack(WcvC1, 128, 96, 128, pCc1);  pack(WcvC2, 96, 64, 96, pCc2);
  pack(WcvK1, 128, 96, 128, pCk1);  pack(WcvK2, 96, 64, 96, pCk2);
  pack(Wo1, 128, 64, 128, pO1);     pack(Wo2, 64, 32, 64, pO2);

  int nagg = N * 64;
  fill_u32_kernel<<<(nagg + 255) / 256, 256, 0, stream>>>(aggC, ENC_NEGINF, nagg);
  fill_u32_kernel<<<(nagg + 255) / 256, 256, 0, stream>>>(aggK, ENC_NEGINF, nagg);

  dim3 blk(256);
  dim3 gN((N + 127) / 128);   // 16 rows/wave * 8 waves/block
  emb_mlp_kernel<<<gN, blk, 0, stream>>>(x_cont, 8, N, pEc1, bc1, pEc2, bc2, pEc3, bc3,
                                         embC32, embC16);
  emb_mlp_kernel<<<gN, blk, 0, stream>>>(x_cat, 3, N, pEk1, bk1, pEk2, bk2, pEk3, bk3,
                                         embK32, embK16);

  dim3 gE((E + 127) / 128);
  edge_conv_kernel<<<gE, blk, 0, stream>>>(embC16, ei, ei + E, pCc1, bcvC1, pCc2, bcvC2,
                                           aggC, E, N);
  edge_conv_kernel<<<gE, blk, 0, stream>>>(embK16, ei, ei + E, pCk1, bcvK1, pCk2, bcvK2,
                                           aggK, E, N);

  out_mlp_kernel<<<gN, blk, 0, stream>>>(embC32, aggC, embK32, aggK,
                                         pO1, bo1, pO2, bo2, Wo3, bo3,
                                         (float*)d_out, N);
}
